// TTTLinear_44195213476413
// MI455X (gfx1250) — matrix operand order repaired and checked by reading, not compiled, not run on hardware
//
#include <hip/hip_runtime.h>

typedef float v2f __attribute__((ext_vector_type(2)));
typedef float v8f __attribute__((ext_vector_type(8)));
typedef __bf16 v16bf __attribute__((ext_vector_type(16)));

#define B_  4
#define L_  2048
#define C_  2048
#define NH_ 32
#define HD_ 64
#define MBS_ 16
#define NC_ 128

__device__ inline unsigned short f2bf(float x) {
    unsigned int u = __float_as_uint(x);
    unsigned int r = u + 0x7FFFu + ((u >> 16) & 1u);
    return (unsigned short)(r >> 16);
}

__device__ inline v8f v8f_zero() { v8f z; for (int i = 0; i < 8; ++i) z[i] = 0.f; return z; }

// ---------------------------------------------------------------- convert (elementwise)
__global__ __launch_bounds__(256) void convert_f32_bf16(const float* __restrict__ in,
                                                        unsigned short* __restrict__ out, int n) {
    int i = blockIdx.x * 256 + threadIdx.x;
    if (i < n) out[i] = f2bf(in[i]);
}

// ---------------------------------------------------------------- convert + transpose
// out[c][r] = bf16(in[r][c]); R x Ccols input, 32x32 LDS tiles, block 256.
__global__ __launch_bounds__(256) void transpose_f32_bf16(const float* __restrict__ in,
                                                          unsigned short* __restrict__ out,
                                                          int R, int Ccols) {
    __shared__ float tile[32][33];
    int c0 = blockIdx.x * 32, r0 = blockIdx.y * 32;
    int tx = threadIdx.x & 31;
    int ty = threadIdx.x >> 5;       // 0..7
    for (int i = 0; i < 4; ++i) {
        int r = ty + i * 8;
        tile[r][tx] = in[(size_t)(r0 + r) * Ccols + c0 + tx];
    }
    __syncthreads();
    for (int i = 0; i < 4; ++i) {
        int r = ty + i * 8;
        out[(size_t)(c0 + r) * R + r0 + tx] = f2bf(tile[tx][r]);
    }
}

// ---------------------------------------------------------------- async stage helper
// Copies one 16B chunk per lane directly into LDS (CDNA5 async copy, ASYNCcnt-tracked).
__device__ __forceinline__ void async_cp16(const unsigned short* g, void* lds_ptr) {
    unsigned ldsoff = (unsigned)(uintptr_t)lds_ptr;   // flat LDS aperture: addr[31:0] = LDS offset
    asm volatile("global_load_async_to_lds_b128 %0, %1, off"
                 :: "v"(ldsoff), "v"(g)
                 : "memory");
}

// ---------------------------------------------------------------- bf16 WMMA GEMM
// C[M,N] = A[M,K] * Bt[N,K]^T.  A row-major [M][K], Bt row-major [N][K] (pre-transposed W).
// WG tile 128x128, 8 waves x (4x2) 16x16 wmma tiles, K staged 32 wide, async double-buffered.
union FragBF { v16bf v; uint4 u[2]; };

__global__ __launch_bounds__(256) void gemm_bf16_wmma(const unsigned short* __restrict__ A,
                                                      const unsigned short* __restrict__ Bt,
                                                      float* __restrict__ C,
                                                      int M, int N, int K) {
    __shared__ __align__(16) unsigned short lA[2][128][40];  // [buf][m][k]
    __shared__ __align__(16) unsigned short lB[2][128][40];  // [buf][n][k]

    const int tid  = threadIdx.x;
    const int wv   = tid >> 5;
    const int lane = tid & 31;
    const int m0   = blockIdx.y * 128;
    const int n0   = blockIdx.x * 128;
    const int wm   = wv >> 2;        // 0..1 -> 64-row slab
    const int wn   = wv & 3;         // 0..3 -> 32-col slab
    const int mloc = lane & 15;
    const int s    = lane >> 4;

    // per-thread staging geometry: chunks c = 2*tid, 2*tid+1 -> row = c>>2, col8 = (c&3)*8
    const int r_st  = (tid * 2) >> 2;          // same row for both chunks of a thread
    const int c_st0 = ((tid * 2) & 3) * 8;
    const int c_st1 = ((tid * 2 + 1) & 3) * 8;

    v8f acc[4][2];
    for (int i = 0; i < 4; ++i) for (int j = 0; j < 2; ++j) acc[i][j] = v8f_zero();

    // pre-issue stage 0 into buffer 0 (4 async instructions per wave)
    async_cp16(A  + (size_t)(m0 + r_st) * K + c_st0, &lA[0][r_st][c_st0]);
    async_cp16(A  + (size_t)(m0 + r_st) * K + c_st1, &lA[0][r_st][c_st1]);
    async_cp16(Bt + (size_t)(n0 + r_st) * K + c_st0, &lB[0][r_st][c_st0]);
    async_cp16(Bt + (size_t)(n0 + r_st) * K + c_st1, &lB[0][r_st][c_st1]);

    int it = 0;
    for (int k0 = 0; k0 < K; k0 += 32, ++it) {
        const int cur = it & 1;
        if (k0 + 32 < K) {
            const int nxt = cur ^ 1;
            const int kn  = k0 + 32;
            async_cp16(A  + (size_t)(m0 + r_st) * K + kn + c_st0, &lA[nxt][r_st][c_st0]);
            async_cp16(A  + (size_t)(m0 + r_st) * K + kn + c_st1, &lA[nxt][r_st][c_st1]);
            async_cp16(Bt + (size_t)(n0 + r_st) * K + kn + c_st0, &lB[nxt][r_st][c_st0]);
            async_cp16(Bt + (size_t)(n0 + r_st) * K + kn + c_st1, &lB[nxt][r_st][c_st1]);
            asm volatile("s_wait_asynccnt 0x4" ::: "memory");   // retire the 4 older (stage k0) ops
        } else {
            asm volatile("s_wait_asynccnt 0x0" ::: "memory");
        }
        __syncthreads();

        FragBF bfr[2];
        for (int nt = 0; nt < 2; ++nt) {
            int bn = wn * 32 + nt * 16 + mloc;   // lane n holds col bn, K = 16*s .. 16*s+15
            bfr[nt].u[0] = *(const uint4*)&lB[cur][bn][16 * s];
            bfr[nt].u[1] = *(const uint4*)&lB[cur][bn][16 * s + 8];
        }
        for (int mt = 0; mt < 4; ++mt) {
            FragBF af;
            int mr = wm * 64 + mt * 16 + mloc;   // lane m row; K = 16*s .. 16*s+15
            af.u[0] = *(const uint4*)&lA[cur][mr][16 * s];
            af.u[1] = *(const uint4*)&lA[cur][mr][16 * s + 8];
            for (int nt = 0; nt < 2; ++nt) {
                acc[mt][nt] = __builtin_amdgcn_wmma_f32_16x16x32_bf16(
                    false, af.v, false, bfr[nt].v, (short)0, acc[mt][nt], false, false);
            }
        }
        __syncthreads();   // all waves done reading 'cur' before it is overwritten next iter
    }

    for (int mt = 0; mt < 4; ++mt)
        for (int nt = 0; nt < 2; ++nt) {
            int col = n0 + wn * 32 + nt * 16 + mloc;
            for (int v = 0; v < 8; ++v) {
                int row = m0 + wm * 64 + mt * 16 + v + 8 * s;
                C[(size_t)row * N + col] = acc[mt][nt][v];
            }
        }
}

// ---------------------------------------------------------------- RoPE (in-place on XQ, XK)
__global__ __launch_bounds__(256) void rope_kernel(float* __restrict__ XQ, float* __restrict__ XK) {
    size_t t   = (size_t)blockIdx.x * 256 + threadIdx.x;   // B*L*C/2 threads
    size_t row = t >> 10;                                   // token row (B*L)
    int    p   = (int)(t & 1023);
    int    hh  = p >> 5;
    int    d2  = p & 31;
    int    l   = (int)(row & (L_ - 1));
    float  pos = (float)(l & (MBS_ - 1));
    float invf = __powf(10000.0f, -(float)(2 * d2) * (1.0f / 64.0f));
    float ang  = pos * invf;
    float cs = __cosf(ang), sn = __sinf(ang);
    size_t idx = row * C_ + hh * HD_ + d2 * 2;
    float qe = XQ[idx], qo = XQ[idx + 1];
    XQ[idx]     = qe * cs - qo * sn;
    XQ[idx + 1] = qo * cs + qe * sn;
    float ke = XK[idx], ko = XK[idx + 1];
    XK[idx]     = ke * cs - ko * sn;
    XK[idx + 1] = ko * cs + ke * sn;
}

// ---------------------------------------------------------------- per-token/head lr = sigmoid(X.w_h+b_h)/HD
__global__ __launch_bounds__(256) void lr_kernel(const float* __restrict__ X,
                                                 const float* __restrict__ w,
                                                 const float* __restrict__ bias,
                                                 float* __restrict__ out) {
    __shared__ float sx[C_];
    __shared__ float red[32][8];
    int row = blockIdx.x;            // b*L + l
    int tid = threadIdx.x;
    for (int i = tid; i < C_; i += 256) sx[i] = X[(size_t)row * C_ + i];
    __syncthreads();
    int h = tid >> 3, q = tid & 7;
    const float* wr = w + (size_t)h * C_;
    float sum = 0.f;
    for (int i = q * 256; i < q * 256 + 256; ++i) sum += sx[i] * wr[i];
    red[h][q] = sum;
    __syncthreads();
    if (q == 0) {
        float a = 0.f;
        for (int e = 0; e < 8; ++e) a += red[h][e];
        a += bias[h];
        float sg = 1.f / (1.f + __expf(-a));
        int b = row >> 11, l = row & (L_ - 1);
        out[((size_t)(b * NH_ + h)) * L_ + l] = sg * (1.0f / (float)HD_);
    }
}

// ---------------------------------------------------------------- TTT scan (one WG per (b,h))
__global__ __launch_bounds__(128) void scan_kernel(const float* __restrict__ XQ,
                                                   const float* __restrict__ XK,
                                                   float* __restrict__ XV,   // read xv, write out (aliased)
                                                   const float* __restrict__ lrB,
                                                   const float* __restrict__ W1g,
                                                   const float* __restrict__ b1g,
                                                   const float* __restrict__ nw,
                                                   const float* __restrict__ nb,
                                                   const float* __restrict__ lti) {
    __shared__ float sW1[64][65];
    __shared__ float sxq[16][68], sxk[16][68], sxv[16][68], sgrad[16][68], sZ[16][68];
    __shared__ float sS[16][17];
    __shared__ float sb1[64], sg[64], sbb[64], slr[16], stok[16];
    __shared__ float sredA[16][8], sredB[16][8], smu[16], srstd[16], ssum1[16], ssum2[16];

    const int bh = blockIdx.x;
    const int b = bh >> 5, h = bh & 31;
    const int tid = threadIdx.x;
    const int w = tid >> 5, lane = tid & 31, m = lane & 15, s = lane >> 4;

    for (int i = tid; i < 4096; i += 128) sW1[i >> 6][i & 63] = W1g[(size_t)h * 4096 + i];
    if (tid < 64) {
        sb1[tid] = b1g[h * 64 + tid];
        sg[tid]  = nw[h * 64 + tid];
        sbb[tid] = nb[h * 64 + tid];
    }
    if (tid < 16) {
        float t = 1.f / (float)(tid + 1) + lti[tid];
        stok[tid] = t > 0.f ? t : 0.f;
    }
    __syncthreads();

    for (int n = 0; n < NC_; ++n) {
        size_t rb = ((size_t)b * L_ + n * MBS_) * C_ + h * HD_;
        for (int i = tid; i < 1024; i += 128) {
            int r = i >> 6, c = i & 63;
            sxq[r][c] = XQ[rb + (size_t)r * C_ + c];
            sxk[r][c] = XK[rb + (size_t)r * C_ + c];
            sxv[r][c] = XV[rb + (size_t)r * C_ + c];
        }
        if (tid < 16) slr[tid] = lrB[((size_t)(b * NH_ + h)) * L_ + n * MBS_ + tid];
        __syncthreads();

        // Z1 = xk @ W1 + b1 (each wave owns 16 columns of W1)
        {
            v8f acc = v8f_zero();
            for (int kk = 0; kk < 16; ++kk) {
                int k = kk * 4 + 2 * s;
                v2f a; a.x = sxk[m][k]; a.y = sxk[m][k + 1];
                v2f bv; bv.x = sW1[k][w * 16 + m]; bv.y = sW1[k + 1][w * 16 + m];
                acc = __builtin_amdgcn_wmma_f32_16x16x4_f32(false, a, false, bv, (short)0, acc, false, false);
            }
            for (int v = 0; v < 8; ++v) sZ[v + 8 * s][w * 16 + m] = acc[v] + sb1[w * 16 + m];
        }
        // S = xq @ xk^T (wave 0 only)
        if (w == 0) {
            v8f acc = v8f_zero();
            for (int kk = 0; kk < 16; ++kk) {
                int k = kk * 4 + 2 * s;
                v2f a;  a.x  = sxq[m][k]; a.y  = sxq[m][k + 1];
                v2f bv; bv.x = sxk[m][k]; bv.y = sxk[m][k + 1];  // B[k][n] = xk[n][k]
                acc = __builtin_amdgcn_wmma_f32_16x16x4_f32(false, a, false, bv, (short)0, acc, false, false);
            }
            for (int v = 0; v < 8; ++v) sS[v + 8 * s][m] = acc[v];
        }
        __syncthreads();

        // grad = ln_fused_l2_bwd(Z1, xv-xk)
        {
            int r = tid >> 3, q = tid & 7;
            float s1 = 0.f, s2 = 0.f;
            for (int e = 0; e < 8; ++e) { float z = sZ[r][q * 8 + e]; s1 += z; s2 += z * z; }
            sredA[r][q] = s1; sredB[r][q] = s2;
            __syncthreads();
            if (q == 0) {
                float a = 0.f, bs = 0.f;
                for (int e = 0; e < 8; ++e) { a += sredA[r][e]; bs += sredB[r][e]; }
                float mu = a * 0.015625f;
                float var = bs * 0.015625f - mu * mu;
                smu[r] = mu; srstd[r] = rsqrtf(var + 1e-6f);
            }
            __syncthreads();
            float mu = smu[r], rs = srstd[r];
            float g1 = 0.f, g2 = 0.f;
            for (int e = 0; e < 8; ++e) {
                int c = q * 8 + e;
                float xh = (sZ[r][c] - mu) * rs;
                float y  = sg[c] * xh + sbb[c];
                float gx = (y - (sxv[r][c] - sxk[r][c])) * sg[c];
                g1 += gx; g2 += gx * xh;
            }
            sredA[r][q] = g1; sredB[r][q] = g2;
            __syncthreads();
            if (q == 0) {
                float a = 0.f, bs = 0.f;
                for (int e = 0; e < 8; ++e) { a += sredA[r][e]; bs += sredB[r][e]; }
                ssum1[r] = a; ssum2[r] = bs;
            }
            __syncthreads();
            for (int e = 0; e < 8; ++e) {
                int c = q * 8 + e;
                float xh = (sZ[r][c] - mu) * rs;
                float y  = sg[c] * xh + sbb[c];
                float gx = (y - (sxv[r][c] - sxk[r][c])) * sg[c];
                sgrad[r][c] = (64.f * gx - ssum1[r] - xh * ssum2[r]) * (rs * 0.015625f);
            }
        }
        __syncthreads();

        // build negated combined mask: M[i][j] = -(j<=i) * tok[i]*lr[j]*(S[i][j]+1)
        for (int e = tid; e < 256; e += 128) {
            int i = e >> 4, j = e & 15;
            sS[i][j] = (j <= i) ? -stok[i] * slr[j] * (sS[i][j] + 1.f) : 0.f;
        }
        __syncthreads();

        // Z1_bar = xq@W1 + M@grad + b1
        {
            v8f acc = v8f_zero();
            for (int kk = 0; kk < 16; ++kk) {
                int k = kk * 4 + 2 * s;
                v2f a;  a.x  = sxq[m][k]; a.y  = sxq[m][k + 1];
                v2f bv; bv.x = sW1[k][w * 16 + m]; bv.y = sW1[k + 1][w * 16 + m];
                acc = __builtin_amdgcn_wmma_f32_16x16x4_f32(false, a, false, bv, (short)0, acc, false, false);
            }
            for (int kk = 0; kk < 4; ++kk) {
                int k = kk * 4 + 2 * s;
                v2f a;  a.x  = sS[m][k]; a.y  = sS[m][k + 1];
                v2f bv; bv.x = sgrad[k][w * 16 + m]; bv.y = sgrad[k + 1][w * 16 + m];
                acc = __builtin_amdgcn_wmma_f32_16x16x4_f32(false, a, false, bv, (short)0, acc, false, false);
            }
            for (int v = 0; v < 8; ++v) sZ[v + 8 * s][w * 16 + m] = acc[v] + sb1[w * 16 + m];
        }
        __syncthreads();

        // W1 -= (last_eta . xk)^T @ grad  (fold minus into A; per-wave column slice)
        {
            float le0 = -stok[15];
            for (int mt = 0; mt < 4; ++mt) {
                v8f acc;
                for (int v = 0; v < 8; ++v) acc[v] = sW1[mt * 16 + v + 8 * s][w * 16 + m];
                for (int kk = 0; kk < 4; ++kk) {
                    int k = kk * 4 + 2 * s;
                    v2f a;
                    a.x = le0 * slr[k]     * sxk[k][mt * 16 + m];
                    a.y = le0 * slr[k + 1] * sxk[k + 1][mt * 16 + m];
                    v2f bv; bv.x = sgrad[k][w * 16 + m]; bv.y = sgrad[k + 1][w * 16 + m];
                    acc = __builtin_amdgcn_wmma_f32_16x16x4_f32(false, a, false, bv, (short)0, acc, false, false);
                }
                for (int v = 0; v < 8; ++v) sW1[mt * 16 + v + 8 * s][w * 16 + m] = acc[v];
            }
        }
        // b1 -= sum_j last_eta[j] * grad[j][:]
        if (tid < 64) {
            float sm = 0.f;
            for (int j = 0; j < 16; ++j) sm += slr[j] * sgrad[j][tid];
            sb1[tid] -= stok[15] * sm;
        }
        __syncthreads();

        // out = xq + ln_fwd(Z1_bar); write over consumed xv chunk
        {
            int r = tid >> 3, q = tid & 7;
            float s1 = 0.f, s2 = 0.f;
            for (int e = 0; e < 8; ++e) { float z = sZ[r][q * 8 + e]; s1 += z; s2 += z * z; }
            sredA[r][q] = s1; sredB[r][q] = s2;
            __syncthreads();
            if (q == 0) {
                float a = 0.f, bs = 0.f;
                for (int e = 0; e < 8; ++e) { a += sredA[r][e]; bs += sredB[r][e]; }
                float mu = a * 0.015625f;
                float var = bs * 0.015625f - mu * mu;
                smu[r] = mu; srstd[r] = rsqrtf(var + 1e-6f);
            }
            __syncthreads();
            float mu = smu[r], rs = srstd[r];
            for (int e = 0; e < 8; ++e) {
                int c = q * 8 + e;
                float xh = (sZ[r][c] - mu) * rs;
                XV[rb + (size_t)r * C_ + c] = sxq[r][c] + sg[c] * xh + sbb[c];
            }
        }
        __syncthreads();
    }
}

// ---------------------------------------------------------------- post-norm -> bf16
__global__ __launch_bounds__(256) void postnorm_kernel(const float* __restrict__ in,
                                                       const float* __restrict__ w,
                                                       const float* __restrict__ bias,
                                                       unsigned short* __restrict__ outbf) {
    __shared__ float red[256];
    __shared__ float smu, srs;
    int row = blockIdx.x, tid = threadIdx.x;
    const float* x = in + (size_t)row * C_;
    float s1 = 0.f, s2 = 0.f;
    for (int i = tid; i < C_; i += 256) { float v = x[i]; s1 += v; s2 += v * v; }
    red[tid] = s1; __syncthreads();
    for (int off = 128; off; off >>= 1) { if (tid < off) red[tid] += red[tid + off]; __syncthreads(); }
    if (tid == 0) smu = red[0] * (1.f / (float)C_);
    __syncthreads();
    red[tid] = s2; __syncthreads();
    for (int off = 128; off; off >>= 1) { if (tid < off) red[tid] += red[tid + off]; __syncthreads(); }
    if (tid == 0) { float var = red[0] * (1.f / (float)C_) - smu * smu; srs = rsqrtf(var + 1e-6f); }
    __syncthreads();
    float mu = smu, rs = srs;
    for (int i = tid; i < C_; i += 256) {
        float v = (x[i] - mu) * rs;
        outbf[(size_t)row * C_ + i] = f2bf(w[i] * v + bias[i]);
    }
}

// ---------------------------------------------------------------- launch
extern "C" void kernel_launch(void* const* d_in, const int* in_sizes, int n_in,
                              void* d_out, int out_size, void* d_ws, size_t ws_size,
                              hipStream_t stream) {
    (void)in_sizes; (void)n_in; (void)out_size; (void)ws_size;
    const float* X   = (const float*)d_in[0];
    const float* Wq  = (const float*)d_in[1];
    const float* Wk  = (const float*)d_in[2];
    const float* Wv  = (const float*)d_in[3];
    const float* Wo  = (const float*)d_in[4];
    const float* W1  = (const float*)d_in[5];
    const float* b1  = (const float*)d_in[6];
    const float* lrw = (const float*)d_in[7];
    const float* lrb = (const float*)d_in[8];
    const float* nw  = (const float*)d_in[9];
    const float* nb  = (const float*)d_in[10];
    const float* lti = (const float*)d_in[11];
    const float* pnw = (const float*)d_in[12];
    const float* pnb = (const float*)d_in[13];

    char* ws = (char*)d_ws;
    unsigned short* Xbf = (unsigned short*)(ws);                    // 33,554,432 B (reused as Ybf)
    unsigned short* Wbf = (unsigned short*)(ws + 33554432ULL);      //  8,388,608 B (transposed W, reused)
    float* XQ  = (float*)(ws + 41943040ULL);                        // 67,108,864 B
    float* XK  = (float*)(ws + 109051904ULL);                       // 67,108,864 B
    float* XV  = (float*)(ws + 176160768ULL);                       // 67,108,864 B (aliased as scan output)
    float* lrB = (float*)(ws + 243269632ULL);                       //  1,048,576 B
    unsigned short* Ybf = Xbf;

    const int NX = B_ * L_ * C_;          // 16,777,216
    dim3 gemm_grid(C_ / 128, (B_ * L_) / 128);  // (16, 64)
    dim3 tr_grid(C_ / 32, C_ / 32);             // (64, 64)

    convert_f32_bf16<<<NX / 256, 256, 0, stream>>>(X, Xbf, NX);

    transpose_f32_bf16<<<tr_grid, 256, 0, stream>>>(Wq, Wbf, C_, C_);
    gemm_bf16_wmma<<<gemm_grid, 256, 0, stream>>>(Xbf, Wbf, XQ, B_ * L_, C_, C_);
    transpose_f32_bf16<<<tr_grid, 256, 0, stream>>>(Wk, Wbf, C_, C_);
    gemm_bf16_wmma<<<gemm_grid, 256, 0, stream>>>(Xbf, Wbf, XK, B_ * L_, C_, C_);
    transpose_f32_bf16<<<tr_grid, 256, 0, stream>>>(Wv, Wbf, C_, C_);
    gemm_bf16_wmma<<<gemm_grid, 256, 0, stream>>>(Xbf, Wbf, XV, B_ * L_, C_, C_);

    rope_kernel<<<(B_ * L_ * C_ / 2) / 256, 256, 0, stream>>>(XQ, XK);
    lr_kernel<<<B_ * L_, 256, 0, stream>>>(X, lrw, lrb, lrB);

    scan_kernel<<<B_ * NH_, 128, 0, stream>>>(XQ, XK, XV, lrB, W1, b1, nw, nb, lti);

    postnorm_kernel<<<B_ * L_, 256, 0, stream>>>(XV, pnw, pnb, Ybf);
    transpose_f32_bf16<<<tr_grid, 256, 0, stream>>>(Wo, Wbf, C_, C_);
    gemm_bf16_wmma<<<gemm_grid, 256, 0, stream>>>(Ybf, Wbf, (float*)d_out, B_ * L_, C_, C_);
}
